// HolisticGradient_50972671869035
// MI455X (gfx1250) — matrix operand order repaired and checked by reading, not compiled, not run on hardware
//
#include <hip/hip_runtime.h>
#include <hip/hip_bf16.h>
#include <math.h>

// ---------------- CDNA5 vector types ----------------
typedef __attribute__((ext_vector_type(16))) __bf16 v16bf;
typedef __attribute__((ext_vector_type(8)))  float  v8f;
typedef __attribute__((ext_vector_type(4)))  unsigned int v4u;

#define EPS 1e-8f

// problem constants
#define Bn 8
#define Cn 128
#define Hn 128
#define Wn 128

// LDS layout (bytes)
#define XP   144                       // Xt pitch in bf16 (row = 288B, 32B multiple)
#define VP   132                       // vol pitch in floats
#define OFF_XT    0                               // 128*144*2      = 36864
#define OFF_VOL   36864                           // 3*128*132*4    = 202752
#define OFF_TAPS  (OFF_VOL + 202752)              // 3*48*4         = 576
#define OFF_XMEAN (OFF_TAPS + 576)                // 3*128*4        = 1536
#define OFF_SPEC  (OFF_XMEAN + 1536)              // 3*128*4        = 1536
#define OFF_MIX   (OFF_SPEC + 1536)               // 128*4          = 512
#define SMEM_BYTES (OFF_MIX + 512)                // = 243776 (< 320KB WGP LDS)

// ---------------- tiny per-batch MLP -> alpha/beta/gamma ----------------
__global__ void hg_mlp_kernel(const float* __restrict__ meta,
                              const float* __restrict__ w1, const float* __restrict__ b1,
                              const float* __restrict__ w2, const float* __restrict__ b2,
                              float* __restrict__ abg) {
    int b = threadIdx.x;
    if (b < Bn) {
        float h[24];
        for (int j = 0; j < 24; ++j) {
            float s = b1[j];
            for (int i = 0; i < 12; ++i) s += meta[b * 12 + i] * w1[j * 12 + i];
            h[j] = fmaxf(s, 0.f);
        }
        for (int k = 0; k < 3; ++k) {         // only alpha,beta,gamma needed
            float s = b2[k];
            for (int j = 0; j < 24; ++j) s += h[j] * w2[k * 24 + j];
            abg[b * 4 + k] = 1.f / (1.f + expf(-s));
        }
    }
}

// K-chunk range of the band for one wave's 16-row c_out block
__device__ __forceinline__ void band_range(int wave, int p, int& kclo, int& kchi) {
    int lo = 16 * wave - p;       if (lo < 0)   lo = 0;
    int hi = 16 * wave + 15 + p;  if (hi > 127) hi = 127;
    kclo = lo >> 5; kchi = hi >> 5;
}

// Build one A fragment (16x32 bf16 banded-kernel tile) for chunk kc.
// ISA 7.12.2 A layout: lane l, elem e -> K = 32*kc + (e&7) + 8*(2*(e>>3) + (l>>4)),
// M = 16*wave + (l&15).  Branch-free gather from zero-sentinel-padded taps.
__device__ __forceinline__ v16bf build_afrag(const float* __restrict__ tapsPad,
                                             int ks, int p, int kc, int M, int lhi) {
    v16bf a;
#pragma unroll
    for (int e = 0; e < 16; ++e) {
        int K = 32 * kc + (e & 7) + 8 * (2 * (e >> 3) + lhi);
        int d = K - M + p;
        int dc = d < -1 ? -1 : (d > ks ? ks : d);
        a[e] = (__bf16)tapsPad[dc + 1];
    }
    return a;
}

// ---------------- banded-conv GEMM via bf16 WMMA ----------------
// D[cout, w] = sum_cin Kmat[cout, cin] * X[cin, w]; A fragments pre-built in VGPRs,
// B = Xt (LDS, [w][c] bf16, contiguous in c -> two ds_load_b128 per fragment).
template <int NSLOT>
__device__ __forceinline__ void band_gemm(const v16bf* __restrict__ af,
                                          int kclo, int kchi,
                                          const __bf16* __restrict__ Xt,
                                          float* __restrict__ volRow,   // nullptr -> no store
                                          float* __restrict__ specAccS, // nullptr -> no acc
                                          int wave, int lane) {
    const int lhi = lane >> 4, llo = lane & 15;

    v8f zero;
#pragma unroll
    for (int q = 0; q < 8; ++q) zero[q] = 0.f;
    v8f acc[8];
#pragma unroll
    for (int n = 0; n < 8; ++n) acc[n] = zero;

#pragma unroll
    for (int slot = 0; slot < NSLOT; ++slot) {
        const int kc = kclo + slot;
        if (kc <= kchi) {                       // wave-uniform guard (EXEC stays full)
#pragma unroll
            for (int n = 0; n < 8; ++n) {
                // B fragment: lane l, elem e -> B[K = 32*kc + (e&7) + 8*(2*(e>>3) + (l>>4)), N = l&15]
                const __bf16* bp = Xt + (16 * n + llo) * XP + 32 * kc + 8 * lhi;
                union { v16bf v; v4u u[2]; } bu;
                const v4u* pp = (const v4u*)bp;
                bu.u[0] = pp[0];
                bu.u[1] = pp[2];
                acc[n] = __builtin_amdgcn_wmma_f32_16x16x32_bf16(
                    false, af[slot], false, bu.v, (short)0, acc[n], false, false);
            }
        }
    }

    // consume: D[M = 16*wave + r + 8*(l>>4), N = 16*n + (l&15)]
#pragma unroll
    for (int n = 0; n < 8; ++n) {
        const int wb = 16 * n + llo;
        float ssum = 0.f;
#pragma unroll
        for (int r8 = 0; r8 < 8; ++r8) {
            float v = fabsf(acc[n][r8]);
            if (volRow) volRow[(16 * wave + r8 + 8 * lhi) * VP + wb] = v;
            ssum += v;
        }
        if (specAccS) atomicAdd(&specAccS[wb], ssum);
    }
}

// ---------------- main kernel: one workgroup per (b,h) row ----------------
__global__ __launch_bounds__(256)
void hg_main_kernel(const float* __restrict__ x, const float* __restrict__ abg,
                    float* __restrict__ out) {
    extern __shared__ char smem[];
    __bf16* Xt      = (__bf16*)(smem + OFF_XT);
    float*  vol     = (float*)(smem + OFF_VOL);
    float*  taps    = (float*)(smem + OFF_TAPS);
    float*  xmean   = (float*)(smem + OFF_XMEAN);
    float*  specAcc = (float*)(smem + OFF_SPEC);
    float*  mixAcc  = (float*)(smem + OFF_MIX);

    const int tid  = threadIdx.x;
    const int wave = tid >> 5, lane = tid & 31;
    const int lhi = lane >> 4, llo = lane & 15;
    const int M = 16 * wave + llo;
    const int b = blockIdx.x >> 7;
    const int h = blockIdx.x & 127;

    const int KS0 = 7, KS1 = 19, KS2 = 43;     // 6*sigma+1 (odd), sigma = 1,3,7
    const int P0 = 3, P1 = 9, P2 = 21;
    const float SIGS[3] = {1.f, 3.f, 7.f};
    const int KSS[3] = {KS0, KS1, KS2};

    // --- init: zero-padded gaussian-derivative taps, zero accumulators ---
    if (tid < 3) {
        const int ks = KSS[tid];
        const float sig = SIGS[tid], s3 = sig * sig * sig, tw = 2.f * sig * sig;
        for (int i = 0; i < 48; ++i) taps[tid * 48 + i] = 0.f;   // sentinels
        float sum = 0.f;
        for (int i = 0; i < ks; ++i) {
            float xx = (float)(i - ks / 2);
            sum += fabsf(-xx / s3 * expf(-xx * xx / tw));
        }
        float inv = 1.f / sum;
        for (int i = 0; i < ks; ++i) {
            float xx = (float)(i - ks / 2);
            taps[tid * 48 + 1 + i] = (-xx / s3 * expf(-xx * xx / tw)) * inv;
        }
    }
    for (int i = tid; i < 4 * Wn; i += 256) {
        if (i < 3 * Wn) specAcc[i] = 0.f; else mixAcc[i - 3 * Wn] = 0.f;
    }
    __syncthreads();

    // --- one-time A-fragment build: banded matrices live in VGPRs all kernel ---
    int kclo0, kchi0, kclo1, kchi1, kclo2, kchi2;
    band_range(wave, P0, kclo0, kchi0);
    band_range(wave, P1, kclo1, kchi1);
    band_range(wave, P2, kclo2, kchi2);
    v16bf af0[2], af1[2], af2[3];
#pragma unroll
    for (int s2 = 0; s2 < 2; ++s2) {
        af0[s2] = build_afrag(taps + 0,  KS0, P0, kclo0 + s2, M, lhi);
        af1[s2] = build_afrag(taps + 48, KS1, P1, kclo1 + s2, M, lhi);
    }
#pragma unroll
    for (int s2 = 0; s2 < 3; ++s2)
        af2[s2] = build_afrag(taps + 96, KS2, P2, kclo2 + s2, M, lhi);

    // --- per neighbor row: stage X row tile, run banded WMMA GEMMs ---
    const float4* x4 = (const float4*)x;
    for (int rr = 0; rr < 3; ++rr) {
        const int row = h + rr - 1;
        const bool inb = (row >= 0) && (row < Hn);   // block-uniform
        __syncthreads();                              // Xt / vol reuse fence
        if (inb) {
            // global (float4, coalesced over w) -> transposed bf16 LDS tile Xt[w][c]
            const bool pf = (rr < 2) && (row + 1 < Hn);
            for (int i = 0; i < (Cn * Wn) / (256 * 4); ++i) {
                int idx = tid + 256 * i;              // 4096 float4 tiles
                int c = idx >> 5, q = idx & 31, w4 = q * 4;
                long gofs = ((long)(b * Cn + c) * Hn + row) * (Wn / 4) + q;
                float4 v = x4[gofs];
                if (pf) __builtin_prefetch(&x4[gofs + (Wn / 4)], 0, 1); // next row -> L2
                Xt[(w4 + 0) * XP + c] = (__bf16)v.x;
                Xt[(w4 + 1) * XP + c] = (__bf16)v.y;
                Xt[(w4 + 2) * XP + c] = (__bf16)v.z;
                Xt[(w4 + 3) * XP + c] = (__bf16)v.w;
            }
        } else {
            float* vr = vol + rr * Cn * VP;
            for (int i = tid; i < Cn * VP; i += 256) vr[i] = 0.f;
            if (tid < Wn) xmean[rr * Wn + tid] = 0.f;
        }
        __syncthreads();
        if (inb) {
            if (tid < Wn) {          // band-mean image row (8 bf16 per ds_load_b128)
                float sm = 0.f;
                const v4u* xr = (const v4u*)(Xt + tid * XP);
                for (int k = 0; k < Cn / 8; ++k) {
                    union { v4u u; __bf16 hh[8]; } cu;
                    cu.u = xr[k];
#pragma unroll
                    for (int j = 0; j < 8; ++j) sm += (float)cu.hh[j];
                }
                xmean[rr * Wn + tid] = sm * (1.f / Cn);
            }
            // scale-1 conv (width 19): vol row always; spectral acc on center
            band_gemm<2>(af1, kclo1, kchi1, Xt, vol + rr * Cn * VP,
                         (rr == 1) ? (specAcc + 1 * Wn) : nullptr, wave, lane);
            if (rr == 1) {           // scales 0 and 2: mean|conv| only
                band_gemm<2>(af0, kclo0, kchi0, Xt, nullptr, specAcc + 0 * Wn, wave, lane);
                band_gemm<3>(af2, kclo2, kchi2, Xt, nullptr, specAcc + 2 * Wn, wave, lane);
            }
        }
    }
    __syncthreads();

    // --- mixed gradient: depthwise Sobel over vol, mean over C via ds atomics ---
    const float SX[3][3] = {{-1,0,1},{-2,0,2},{-1,0,1}};
    const float SY[3][3] = {{-1,-2,-1},{0,0,0},{1,2,1}};
    for (int i = tid; i < Cn * Wn; i += 256) {
        int c = i >> 7, w = i & 127;
        float mx = 0.f, my = 0.f;
        for (int rr = 0; rr < 3; ++rr) {
            const float* vr = vol + (rr * Cn + c) * VP;
            for (int j = 0; j < 3; ++j) {
                int ww = w + j - 1;
                if (ww >= 0 && ww < Wn) {
                    float v = vr[ww];
                    mx += v * SX[rr][j];
                    my += v * SY[rr][j];
                }
            }
        }
        atomicAdd(&mixAcc[w], sqrtf(mx * mx + my * my + EPS));
    }
    __syncthreads();

    // --- final combine ---
    if (tid < Wn) {
        const int w = tid;
        float gx = 0.f, gy = 0.f;
        for (int rr = 0; rr < 3; ++rr)
            for (int j = 0; j < 3; ++j) {
                int ww = w + j - 1;
                if (ww >= 0 && ww < Wn) {
                    float v = xmean[rr * Wn + ww];
                    gx += v * SX[rr][j];
                    gy += v * SY[rr][j];
                }
            }
        float sg2 = gx * gx + gy * gy + EPS;              // spatial_grad^2
        float sp = fmaxf(specAcc[w], fmaxf(specAcc[Wn + w], specAcc[2 * Wn + w])) * (1.f / Cn);
        float mg = mixAcc[w] * (1.f / Cn);
        float a_ = abg[b * 4 + 0], b_ = abg[b * 4 + 1], g_ = abg[b * 4 + 2];
        out[(b * Hn + h) * Wn + w] =
            sqrtf(a_ * sg2 + b_ * sp * sp + g_ * mg * mg + EPS);
    }
}

// ---------------- launcher ----------------
extern "C" void kernel_launch(void* const* d_in, const int* in_sizes, int n_in,
                              void* d_out, int out_size, void* d_ws, size_t ws_size,
                              hipStream_t stream) {
    const float* x    = (const float*)d_in[0];
    const float* meta = (const float*)d_in[1];
    const float* w1   = (const float*)d_in[2];
    const float* b1   = (const float*)d_in[3];
    const float* w2   = (const float*)d_in[4];
    const float* b2   = (const float*)d_in[5];
    float* out = (float*)d_out;
    float* abg = (float*)d_ws;       // 8*4 floats

    hg_mlp_kernel<<<1, 32, 0, stream>>>(meta, w1, b1, w2, b2, abg);
    hg_main_kernel<<<Bn * Hn, 256, SMEM_BYTES, stream>>>(x, abg, out);
}
